// MultiheadCrossAttention_65910568124478
// MI455X (gfx1250) — compile-verified, matrix operand order checked by reading, not run on hardware
//
#include <hip/hip_runtime.h>
#include <hip/hip_bf16.h>

typedef __attribute__((ext_vector_type(16))) _Float16 v16h;
typedef __attribute__((ext_vector_type(8)))  _Float16 v8h;
typedef __attribute__((ext_vector_type(8)))  float    v8f;

#define BB 8
#define SS 1024
#define FF 512
#define EE 64
#define HH 16
#define MM (BB*SS)      // 8192 rows
#define NKV (EE*HH*2)   // 2048
#define NQ  (EE*HH)     // 1024

// ---------------------------------------------------------------------------
// Fragment load: 16-bit A/B operand, 16x32 tile, row-major source.
// lane (0-15): row = lane, K = {0..7, 16..23}; lane (16-31): row = lane-16,
// K = {8..15, 24..31}.  koff_half = (lane>>4)*8.  Two global_load_b128.
// ---------------------------------------------------------------------------
__device__ __forceinline__ v16h load_frag16(const _Float16* __restrict__ rowptr,
                                            int koff_half) {
  v8h lo = *(const v8h*)(rowptr + koff_half);
  v8h hi = *(const v8h*)(rowptr + koff_half + 16);
  v16h r;
#pragma unroll
  for (int i = 0; i < 8; ++i) { r[i] = lo[i]; r[8 + i] = hi[i]; }
  return r;
}

__device__ __forceinline__ v8f wmma_f16(v16h a, v16h b, v8f c) {
  return __builtin_amdgcn_wmma_f32_16x16x32_f16(
      /*neg_a=*/false, a, /*neg_b=*/false, b,
      /*c_mod=*/(short)0, c, /*reuse_a=*/false, /*reuse_b=*/false);
}

// ---------------------------------------------------------------------------
// Elementwise f32 -> f16 convert (activations)
// ---------------------------------------------------------------------------
__global__ void cvt_x_kernel(const float* __restrict__ src,
                             _Float16* __restrict__ dst, int n) {
  int i = blockIdx.x * blockDim.x + threadIdx.x;
  if (i < n) dst[i] = (_Float16)src[i];
}

// W[k][n] (K x N, f32) -> Wt[n][k] (N x K, f16)
__global__ void cvt_wT_kernel(const float* __restrict__ w,
                              _Float16* __restrict__ wT, int K, int N) {
  int i = blockIdx.x * blockDim.x + threadIdx.x;
  if (i < N * K) {
    int n = i / K, k = i % K;
    wT[i] = (_Float16)w[(size_t)k * N + n];
  }
}

// ---------------------------------------------------------------------------
// q GEMM: (8192 x 512) x (512 x 1024) + b_q, scatter to Q[b][h][s][e] f16.
// Each wave: 16(M) x 64(N) tile, K-loop in steps of 32 -> 64 v_wmma per wave.
// ---------------------------------------------------------------------------
__global__ void gemm_q_kernel(const _Float16* __restrict__ xh,
                              const _Float16* __restrict__ wT,
                              const float* __restrict__ bias,
                              _Float16* __restrict__ q_ws) {
  const int lane = threadIdx.x & 31;
  const int wave = blockIdx.x * (blockDim.x >> 5) + (threadIdx.x >> 5);
  const int NT64 = NQ / 64;              // 16
  const int mt  = wave / NT64;           // 0..511
  const int n64 = wave % NT64;
  const int col = lane & 15;
  const int koff = (lane >> 4) * 8;

  const _Float16* arow = xh + (size_t)(mt * 16 + col) * FF;
  v8f acc[4] = {};
  for (int k = 0; k < FF; k += 32) {
    v16h a = load_frag16(arow + k, koff);
#pragma unroll
    for (int j = 0; j < 4; ++j) {
      int n = n64 * 64 + j * 16 + col;
      v16h bf = load_frag16(wT + (size_t)n * FF + k, koff);
      acc[j] = wmma_f16(a, bf, acc[j]);
    }
  }
#pragma unroll
  for (int j = 0; j < 4; ++j) {
    int n = n64 * 64 + j * 16 + col;   // n = e*16 + h
    float bv = bias[n];
    int e = n >> 4, h = n & 15;
#pragma unroll
    for (int r = 0; r < 8; ++r) {
      int m = mt * 16 + (lane >> 4) * 8 + r;
      int b = m >> 10, s = m & (SS - 1);
      q_ws[(((size_t)(b * HH + h) * SS + s) * EE) + e] =
          (_Float16)(acc[j][r] + bv);
    }
  }
}

// ---------------------------------------------------------------------------
// kv GEMM: (8192 x 512) x (512 x 2048) + b_kv, scatter:
//   c==0 -> K[b][h][s][e]      (row-major for QK^T A-operand loads)
//   c==1 -> Vt[b][h][e][s]     (transposed: row-major for PV A-operand loads)
// ---------------------------------------------------------------------------
__global__ void gemm_kv_kernel(const _Float16* __restrict__ xh,
                               const _Float16* __restrict__ wT,
                               const float* __restrict__ bias,
                               _Float16* __restrict__ k_ws,
                               _Float16* __restrict__ vT_ws) {
  const int lane = threadIdx.x & 31;
  const int wave = blockIdx.x * (blockDim.x >> 5) + (threadIdx.x >> 5);
  const int NT64 = NKV / 64;             // 32
  const int mt  = wave / NT64;           // 0..511
  const int n64 = wave % NT64;
  const int col = lane & 15;
  const int koff = (lane >> 4) * 8;

  const _Float16* arow = xh + (size_t)(mt * 16 + col) * FF;
  v8f acc[4] = {};
  for (int k = 0; k < FF; k += 32) {
    v16h a = load_frag16(arow + k, koff);
#pragma unroll
    for (int j = 0; j < 4; ++j) {
      int n = n64 * 64 + j * 16 + col;
      v16h bf = load_frag16(wT + (size_t)n * FF + k, koff);
      acc[j] = wmma_f16(a, bf, acc[j]);
    }
  }
#pragma unroll
  for (int j = 0; j < 4; ++j) {
    int n = n64 * 64 + j * 16 + col;   // n = (e*16 + h)*2 + c
    float bv = bias[n];
    int e = n >> 5, h = (n >> 1) & 15, c = n & 1;
#pragma unroll
    for (int r = 0; r < 8; ++r) {
      int m = mt * 16 + (lane >> 4) * 8 + r;
      int b = m >> 10, s = m & (SS - 1);
      _Float16 val = (_Float16)(acc[j][r] + bv);
      size_t bh = (size_t)(b * HH + h);
      if (c == 0)
        k_ws[(bh * SS + s) * EE + e] = val;
      else
        vT_ws[(bh * EE + e) * SS + s] = val;
    }
  }
}

// ---------------------------------------------------------------------------
// Fused flash attention, one wave per (b,h, 16 query rows).
//   S^T = K_tile(16k x 64e) . Q^T            (A = K rows, B = Q rows)
//   online softmax along k (row stat lives on lane = q, one xor16 exchange)
//   O^T += V^T(16e x 32k) . P^T              (P^T frag built in-lane from S^T)
// Final: out[b][s][h*64+e] = O^T * (0.125 / l)
// ---------------------------------------------------------------------------
__global__ void attn_kernel(const _Float16* __restrict__ q_ws,
                            const _Float16* __restrict__ k_ws,
                            const _Float16* __restrict__ vT_ws,
                            float* __restrict__ out) {
  const int lane = threadIdx.x & 31;
  const int wave = blockIdx.x * (blockDim.x >> 5) + (threadIdx.x >> 5);
  const int bh = wave >> 6;        // 0..127
  const int qt = wave & 63;        // 16-row query tile
  const int col = lane & 15;
  const int koff = (lane >> 4) * 8;

  const _Float16* Qb = q_ws + ((size_t)bh * SS + qt * 16) * EE;
  const _Float16* Kb = k_ws + (size_t)bh * SS * EE;
  const _Float16* Vt = vT_ws + (size_t)bh * EE * SS;

  // Q as B-operand: lane col = query row, contraction over e.
  const _Float16* qrow = Qb + (size_t)col * EE;
  v16h qf0 = load_frag16(qrow + 0,  koff);
  v16h qf1 = load_frag16(qrow + 32, koff);

  v8f o[4] = {};
  float mrow = -1e30f, lrow = 0.0f;

  for (int kt = 0; kt < SS; kt += 32) {
    const _Float16* kr0 = Kb + (size_t)(kt + col) * EE;
    const _Float16* kr1 = Kb + (size_t)(kt + 16 + col) * EE;
    v16h ka0 = load_frag16(kr0 + 0,  koff);
    v16h ka1 = load_frag16(kr0 + 32, koff);
    v16h kb0 = load_frag16(kr1 + 0,  koff);
    v16h kb1 = load_frag16(kr1 + 32, koff);

    v8f s0 = {}, s1 = {};
    s0 = wmma_f16(ka0, qf0, s0);  s0 = wmma_f16(ka1, qf1, s0);  // k = kt+0..15
    s1 = wmma_f16(kb0, qf0, s1);  s1 = wmma_f16(kb1, qf1, s1);  // k = kt+16..31

    // chunk max over the 32 keys for query row = col
    float cm = s0[0];
#pragma unroll
    for (int r = 0; r < 8; ++r) { cm = fmaxf(cm, s0[r]); cm = fmaxf(cm, s1[r]); }
    cm = fmaxf(cm, __shfl_xor(cm, 16, 32));
    float mnew = fmaxf(mrow, cm);
    float alpha = __expf(mrow - mnew);

    float p0[8], p1[8], ps = 0.0f;
#pragma unroll
    for (int r = 0; r < 8; ++r) {
      p0[r] = __expf(s0[r] - mnew); ps += p0[r];
      p1[r] = __expf(s1[r] - mnew); ps += p1[r];
    }
    ps += __shfl_xor(ps, 16, 32);
    lrow = lrow * alpha + ps;
    mrow = mnew;

#pragma unroll
    for (int f = 0; f < 4; ++f)
#pragma unroll
      for (int r = 0; r < 8; ++r) o[f][r] *= alpha;

    // S^T C-layout is already the B-operand layout: pack f32->f16 in-lane.
    v16h pf;
#pragma unroll
    for (int i = 0; i < 8; ++i) {
      pf[i]     = (_Float16)p0[i];
      pf[8 + i] = (_Float16)p1[i];
    }

#pragma unroll
    for (int f = 0; f < 4; ++f) {
      const _Float16* vrow = Vt + (size_t)(f * 16 + col) * SS + kt;
      v16h vf = load_frag16(vrow, koff);
      o[f] = wmma_f16(vf, pf, o[f]);   // O^T[e][q] accumulate
    }
  }

  const float scale = 0.125f / lrow;   // 1/sqrt(E) applied post-softmax
  const int qrow_g = qt * 16 + col;
  const int b = bh >> 4, h = bh & 15;
  float* ob = out + ((size_t)(b * SS + qrow_g)) * (EE * HH) + h * EE;
#pragma unroll
  for (int f = 0; f < 4; ++f)
#pragma unroll
    for (int r = 0; r < 8; ++r)
      ob[f * 16 + (lane >> 4) * 8 + r] = o[f][r] * scale;
}

// ---------------------------------------------------------------------------
extern "C" void kernel_launch(void* const* d_in, const int* in_sizes, int n_in,
                              void* d_out, int out_size, void* d_ws, size_t ws_size,
                              hipStream_t stream) {
  (void)in_sizes; (void)n_in; (void)out_size; (void)ws_size;
  const float* inputs1 = (const float*)d_in[0];
  const float* inputs2 = (const float*)d_in[1];
  const float* W_kv    = (const float*)d_in[2];
  const float* b_kv    = (const float*)d_in[3];
  const float* W_q     = (const float*)d_in[4];
  const float* b_q     = (const float*)d_in[5];
  float* out = (float*)d_out;

  char* p = (char*)d_ws;
  _Float16* x1h   = (_Float16*)p; p += (size_t)MM * FF * 2;        // 8 MB
  _Float16* x2h   = (_Float16*)p; p += (size_t)MM * FF * 2;        // 8 MB
  _Float16* wkvT  = (_Float16*)p; p += (size_t)NKV * FF * 2;       // 2 MB
  _Float16* wqT   = (_Float16*)p; p += (size_t)NQ  * FF * 2;       // 1 MB
  _Float16* q_ws  = (_Float16*)p; p += (size_t)BB * HH * SS * EE * 2; // 16 MB
  _Float16* k_ws  = (_Float16*)p; p += (size_t)BB * HH * SS * EE * 2; // 16 MB
  _Float16* vT_ws = (_Float16*)p;                                      // 16 MB

  const int thr = 256;
  cvt_x_kernel<<<(MM * FF + thr - 1) / thr, thr, 0, stream>>>(inputs1, x1h, MM * FF);
  cvt_x_kernel<<<(MM * FF + thr - 1) / thr, thr, 0, stream>>>(inputs2, x2h, MM * FF);
  cvt_wT_kernel<<<(NKV * FF + thr - 1) / thr, thr, 0, stream>>>(W_kv, wkvT, FF, NKV);
  cvt_wT_kernel<<<(NQ  * FF + thr - 1) / thr, thr, 0, stream>>>(W_q,  wqT,  FF, NQ);

  // kv GEMM: (8192/16) * (2048/64) = 16384 waves -> 2048 blocks of 8 waves
  gemm_kv_kernel<<<2048, 256, 0, stream>>>(x1h, wkvT, b_kv, k_ws, vT_ws);
  // q GEMM: 512 * 16 = 8192 waves -> 1024 blocks
  gemm_q_kernel<<<1024, 256, 0, stream>>>(x2h, wqT, b_q, q_ws);
  // attention: 128 (b,h) * 64 q-tiles = 8192 waves -> 1024 blocks
  attn_kernel<<<1024, 256, 0, stream>>>(q_ws, k_ws, vT_ws, out);
}
